// RecursiveCompressionLayer_54915451847058
// MI455X (gfx1250) — matrix operand order repaired and checked by reading, not compiled, not run on hardware
//
#include <hip/hip_runtime.h>
#include <hip/hip_bf16.h>
#include <math.h>

typedef __bf16 bf16;
typedef __attribute__((ext_vector_type(16))) __bf16 v16bf;
typedef __attribute__((ext_vector_type(8)))  float   v8f;

// Problem constants (D=1024, B=4, S=1024)
#define Dd   1024
#define Ss   1024
#define Bb   4
#define Tt   4096          // B*S
#define Ff   7
#define CDm  512           // CDMAX
#define Hh   512           // D/2
#define Rr   256           // D/4
#define NBINS 4096

// GEMM tiling: 64(M)x128(N) block tile, K-step 32, 256 threads = 8 wave32.
// Each wave owns a 32x32 output tile => 4 WMMAs per K-step from 2 A + 2 B frags.
#define BM 64
#define BN 128
#define BK 32
#define LDK 34             // padded LDS row stride (17 dwords, coprime with 64 banks)

// ---------------------------------------------------------------------------
// WMMA fragment loaders (wave32); contiguous k order, same for A and B.
// A (16x32 bf16): lane holds row=lane&15; lane<16 K=0..15, lane>=16 K=16..31
// B (32x16 bf16): lane holds col=lane&15; lane<16 K=0..15, lane>=16 K=16..31
// LDS tiles are stored K-contiguous: A as [M][K], B as [N][K] (i.e. W^T tile).
// ---------------------------------------------------------------------------
__device__ __forceinline__ v16bf frag_A16x32(const bf16* tile, int lane) {
  const int r  = lane & 15;
  const int kh = (lane >> 4) << 4;           // 0 or 16
  const bf16* p = tile + r * LDK;
  v16bf f;
#pragma unroll
  for (int j = 0; j < 16; ++j) f[j] = p[kh + j];
  return f;
}

__device__ __forceinline__ v16bf frag_B32x16(const bf16* tileNT, int lane) {
  const int n  = lane & 15;
  const int kh = (lane >> 4) << 4;           // 0 or 16
  const bf16* p = tileNT + n * LDK + kh;
  v16bf f;
#pragma unroll
  for (int j = 0; j < 16; ++j) f[j] = p[j];
  return f;
}

// 2x2 register-tile WMMA step: wave covers rows wm*32..+31, cols wn*32..+31
__device__ __forceinline__ void wmma_step(const bf16* As, const bf16* Bs,
                                          int wm, int wn, int lane,
                                          v8f acc[2][2]) {
  v16bf a0 = frag_A16x32(As + (wm * 32) * LDK, lane);
  v16bf a1 = frag_A16x32(As + (wm * 32 + 16) * LDK, lane);
  v16bf b0 = frag_B32x16(Bs + (wn * 32) * LDK, lane);
  v16bf b1 = frag_B32x16(Bs + (wn * 32 + 16) * LDK, lane);
  acc[0][0] = __builtin_amdgcn_wmma_f32_16x16x32_bf16(false, a0, false, b0, (short)0, acc[0][0], false, false);
  acc[0][1] = __builtin_amdgcn_wmma_f32_16x16x32_bf16(false, a0, false, b1, (short)0, acc[0][1], false, false);
  acc[1][0] = __builtin_amdgcn_wmma_f32_16x16x32_bf16(false, a1, false, b0, (short)0, acc[1][0], false, false);
  acc[1][1] = __builtin_amdgcn_wmma_f32_16x16x32_bf16(false, a1, false, b1, (short)0, acc[1][1], false, false);
}

template<int ACT>
__device__ __forceinline__ float activate(float v) {
  if (ACT == 1) v = 0.5f * v * (1.0f + erff(v * 0.70710678118654752f)); // exact GELU
  if (ACT == 2) v = 1.0f / (1.0f + expf(-v));                          // sigmoid
  return v;
}

// ---------------------------------------------------------------------------
// Generic dense GEMM: out = act(A[M,K]@W[K,N] + bias) (+add0 +add1), bf16 WMMA
// ---------------------------------------------------------------------------
template<int ACT>
__global__ __launch_bounds__(256) void gemm_kernel(
    const bf16* __restrict__ A, const bf16* __restrict__ W,
    const float* __restrict__ bias,
    float* __restrict__ outF, bf16* __restrict__ outB,
    const float* __restrict__ add0, const float* __restrict__ add1,
    int M, int N, int K)
{
  __shared__ bf16 As[BM * LDK];
  __shared__ bf16 Bs[BN * LDK];
  const int tid = threadIdx.x, lane = tid & 31, wave = tid >> 5;
  const int m0 = blockIdx.x * BM, n0 = blockIdx.y * BN;
  const int wm = wave >> 2, wn = wave & 3;
  v8f acc[2][2] = {};
  for (int k0 = 0; k0 < K; k0 += BK) {
    __syncthreads();
    for (int i = tid; i < BM * (BK / 2); i += 256) {
      int r = i / (BK / 2), kp = (i % (BK / 2)) * 2;
      int gr = m0 + r;
      unsigned v = 0;
      if (gr < M) v = *(const unsigned*)(A + (size_t)gr * K + k0 + kp);
      *(unsigned*)(As + r * LDK + kp) = v;
    }
    for (int i = tid; i < BK * (BN / 2); i += 256) {
      int k = i / (BN / 2), np = (i % (BN / 2)) * 2;
      union { unsigned u; bf16 h[2]; } cv;
      cv.u = *(const unsigned*)(W + (size_t)(k0 + k) * N + n0 + np);
      Bs[np * LDK + k]       = cv.h[0];
      Bs[(np + 1) * LDK + k] = cv.h[1];
    }
    // speculative prefetch of next K tile (global_prefetch_b8, OOB-safe)
    if (k0 + BK < K) {
      int pr = m0 + (tid & (BM - 1));
      if (pr < M) __builtin_prefetch(A + (size_t)pr * K + k0 + BK, 0, 1);
      __builtin_prefetch(W + (size_t)(k0 + BK + (tid & (BK - 1))) * N + n0, 0, 1);
    }
    __syncthreads();
    wmma_step(As, Bs, wm, wn, lane, acc);
  }
  const int col = lane & 15, rb = (lane >> 4) * 8;
#pragma unroll
  for (int ai = 0; ai < 2; ++ai) {
#pragma unroll
    for (int bj = 0; bj < 2; ++bj) {
      int nn = n0 + wn * 32 + bj * 16 + col;
#pragma unroll
      for (int i = 0; i < 8; ++i) {
        int gr = m0 + wm * 32 + ai * 16 + rb + i;
        if (gr < M) {
          float v = acc[ai][bj][i] + (bias ? bias[nn] : 0.0f);
          v = activate<ACT>(v);
          size_t oi = (size_t)gr * N + nn;
          if (add0) v += add0[oi];
          if (add1) v += add1[oi];
          if (outF) outF[oi] = v;
          if (outB) outB[oi] = (bf16)v;
        }
      }
    }
  }
}

// ---------------------------------------------------------------------------
// Routed-expert GEMM: rows are one expert's compacted segment. GATHER reads A
// rows through perm[] (tokens); otherwise A is the compact intermediate buffer.
// Output is always compact. Weights/bias indexed per expert (blockIdx.z).
// ---------------------------------------------------------------------------
template<int ACT, bool GATHER>
__global__ __launch_bounds__(256) void expert_gemm_kernel(
    const bf16* __restrict__ A, const bf16* __restrict__ Wbase,
    const float* __restrict__ bbase,
    float* __restrict__ outF, bf16* __restrict__ outB,
    const int* __restrict__ segOff, const int* __restrict__ perm,
    int N, int K, long wStride, int bStride)
{
  const int e    = blockIdx.z;
  const int base = segOff[e];
  const int Me   = segOff[e + 1] - base;
  const int m0   = blockIdx.x * BM;
  if (m0 >= Me) return;
  const bf16*  W    = Wbase + (size_t)e * wStride;
  const float* bias = bbase + (size_t)e * bStride;

  __shared__ bf16 As[BM * LDK];
  __shared__ bf16 Bs[BN * LDK];
  const int tid = threadIdx.x, lane = tid & 31, wave = tid >> 5;
  const int n0 = blockIdx.y * BN;
  const int wm = wave >> 2, wn = wave & 3;
  v8f acc[2][2] = {};
  for (int k0 = 0; k0 < K; k0 += BK) {
    __syncthreads();
    for (int i = tid; i < BM * (BK / 2); i += 256) {
      int r = i / (BK / 2), kp = (i % (BK / 2)) * 2;
      int gr = m0 + r;
      unsigned v = 0;
      if (gr < Me) {
        int ar = GATHER ? perm[base + gr] : (base + gr);
        v = *(const unsigned*)(A + (size_t)ar * K + k0 + kp);
      }
      *(unsigned*)(As + r * LDK + kp) = v;
    }
    for (int i = tid; i < BK * (BN / 2); i += 256) {
      int k = i / (BN / 2), np = (i % (BN / 2)) * 2;
      union { unsigned u; bf16 h[2]; } cv;
      cv.u = *(const unsigned*)(W + (size_t)(k0 + k) * N + n0 + np);
      Bs[np * LDK + k]       = cv.h[0];
      Bs[(np + 1) * LDK + k] = cv.h[1];
    }
    if (k0 + BK < K)
      __builtin_prefetch(W + (size_t)(k0 + BK + (tid & (BK - 1))) * N + n0, 0, 1);
    __syncthreads();
    wmma_step(As, Bs, wm, wn, lane, acc);
  }
  const int col = lane & 15, rb = (lane >> 4) * 8;
#pragma unroll
  for (int ai = 0; ai < 2; ++ai) {
#pragma unroll
    for (int bj = 0; bj < 2; ++bj) {
      int nn = n0 + wn * 32 + bj * 16 + col;
#pragma unroll
      for (int i = 0; i < 8; ++i) {
        int gr = m0 + wm * 32 + ai * 16 + rb + i;
        if (gr < Me) {
          float v = acc[ai][bj][i] + bias[nn];
          v = activate<ACT>(v);
          size_t oi = (size_t)(base + gr) * N + nn;
          if (outF) outF[oi] = v;
          if (outB) outB[oi] = (bf16)v;
        }
      }
    }
  }
}

// ---------------------------------------------------------------------------
// Pairwise cosine sims: S = nrm @ nrm^T (4096x4096, K=1024). Epilogue builds a
// 4096-bin LDS histogram (for the 95th percentile) and stores same-batch
// diagonal blocks to sim_b for the redundancy pass.
// ---------------------------------------------------------------------------
__global__ __launch_bounds__(256) void sims_kernel(
    const bf16* __restrict__ nrm, float* __restrict__ sim_b,
    unsigned* __restrict__ hist)
{
  __shared__ bf16 As[BM * LDK];
  __shared__ bf16 Bs[BN * LDK];
  __shared__ unsigned lh[NBINS];
  const int tid = threadIdx.x, lane = tid & 31, wave = tid >> 5;
  const int m0 = blockIdx.x * BM, n0 = blockIdx.y * BN;
  const int wm = wave >> 2, wn = wave & 3;
  for (int i = tid; i < NBINS; i += 256) lh[i] = 0;
  v8f acc[2][2] = {};
  for (int k0 = 0; k0 < Dd; k0 += BK) {
    __syncthreads();
    for (int i = tid; i < BM * (BK / 2); i += 256) {
      int r = i / (BK / 2), kp = (i % (BK / 2)) * 2;
      *(unsigned*)(As + r * LDK + kp) =
          *(const unsigned*)(nrm + (size_t)(m0 + r) * Dd + k0 + kp);
    }
    for (int i = tid; i < BN * (BK / 2); i += 256) {
      int r = i / (BK / 2), kp = (i % (BK / 2)) * 2;
      *(unsigned*)(Bs + r * LDK + kp) =
          *(const unsigned*)(nrm + (size_t)(n0 + r) * Dd + k0 + kp);
    }
    if (k0 + BK < Dd) {
      __builtin_prefetch(nrm + (size_t)(m0 + (tid & (BM - 1))) * Dd + k0 + BK, 0, 1);
      __builtin_prefetch(nrm + (size_t)(n0 + (tid & (BN - 1))) * Dd + k0 + BK, 0, 1);
    }
    __syncthreads();
    wmma_step(As, Bs, wm, wn, lane, acc);   // B tile already [N][K]
  }
  __syncthreads();
  const int col = lane & 15, rb = (lane >> 4) * 8;
  const int bi = m0 >> 10, bj = n0 >> 10;   // batch of the tile (64|1024, 128|1024)
#pragma unroll
  for (int ai = 0; ai < 2; ++ai) {
#pragma unroll
    for (int bjx = 0; bjx < 2; ++bjx) {
      int gj = n0 + wn * 32 + bjx * 16 + col;
#pragma unroll
      for (int i = 0; i < 8; ++i) {
        int gi = m0 + wm * 32 + ai * 16 + rb + i;
        float s = acc[ai][bjx][i];
        int bin = (int)((s + 1.0f) * (0.5f * NBINS));
        bin = bin < 0 ? 0 : (bin > NBINS - 1 ? NBINS - 1 : bin);
        atomicAdd(&lh[bin], 1u);
        if (bi == bj)
          sim_b[((size_t)bi * Ss + (gi & (Ss - 1))) * Ss + (gj & (Ss - 1))] = s;
      }
    }
  }
  __syncthreads();
  for (int i = tid; i < NBINS; i += 256) {
    unsigned c = lh[i];
    if (c) atomicAdd(&hist[i], c);
  }
}

// ------------------------- small support kernels ---------------------------
__global__ void f2bf_kernel(const float* __restrict__ s, bf16* __restrict__ d,
                            long long n) {
  long long i = (long long)blockIdx.x * blockDim.x + threadIdx.x;
  if (i < n) d[i] = (bf16)s[i];
}

// per-token L2 norm; emits bf16 tokens and bf16 normalized tokens
__global__ __launch_bounds__(256) void norm_kernel(
    const float* __restrict__ tokens, bf16* __restrict__ tok_bf,
    bf16* __restrict__ nrm_bf) {
  const int t = blockIdx.x, tid = threadIdx.x;
  const float* row = tokens + (size_t)t * Dd;
  __shared__ float sred[256];
  float ss = 0.0f;
  for (int d = tid; d < Dd; d += 256) { float v = row[d]; ss += v * v; }
  sred[tid] = ss; __syncthreads();
  for (int s = 128; s > 0; s >>= 1) { if (tid < s) sred[tid] += sred[tid + s]; __syncthreads(); }
  const float scale = 1.0f / (sqrtf(sred[0]) + 1e-12f);
  for (int d = tid; d < Dd; d += 256) {
    float v = row[d];
    tok_bf[(size_t)t * Dd + d] = (bf16)v;
    nrm_bf[(size_t)t * Dd + d] = (bf16)(v * scale);
  }
}

__global__ void ctx_kernel(const float* __restrict__ tokens, float* __restrict__ ctx) {
  const int b = blockIdx.y;
  const int d = blockIdx.x * blockDim.x + threadIdx.x;
  if (d < Dd) {
    const float* p = tokens + (size_t)b * Ss * Dd + d;
    float s = 0.0f;
    for (int t = 0; t < Ss; ++t) s += p[(size_t)t * Dd];
    ctx[b * Dd + d] = s * (1.0f / Ss);
  }
}

// histogram quantile + per-batch adaptive threshold
__global__ __launch_bounds__(256) void thr_kernel(
    const unsigned* __restrict__ hist, const float* __restrict__ ctx,
    const float* __restrict__ w_tm, const float* __restrict__ b_tm,
    float* __restrict__ thr) {
  __shared__ float q_s;
  __shared__ float sred[256];
  const int tid = threadIdx.x;
  if (tid == 0) {
    unsigned long long total = 0;
    for (int i = 0; i < NBINS; ++i) total += hist[i];
    double target = 0.95 * (double)(total - 1);
    unsigned long long cum = 0; int bin = NBINS - 1;
    for (int i = 0; i < NBINS; ++i) {
      cum += hist[i];
      if ((double)cum >= target + 1.0) { bin = i; break; }
    }
    q_s = -1.0f + (bin + 0.5f) * (2.0f / NBINS);
  }
  __syncthreads();
  for (int b = 0; b < Bb; ++b) {
    float p = 0.0f;
    for (int d = tid; d < Dd; d += 256) p += ctx[b * Dd + d] * w_tm[d];
    sred[tid] = p; __syncthreads();
    for (int s = 128; s > 0; s >>= 1) { if (tid < s) sred[tid] += sred[tid + s]; __syncthreads(); }
    if (tid == 0) {
      float mod = (1.0f / (1.0f + expf(-(sred[0] + b_tm[0])))) * 0.1f;
      float t = 0.75f + 0.25f * q_s + mod;
      thr[b] = fminf(fmaxf(t, 0.75f), 0.98f);
    }
    __syncthreads();
  }
}

// redundancy[b,s] = mean_j( sim*pen > thr[b] ); pen applied on the fly
__global__ __launch_bounds__(256) void red_kernel(
    const float* __restrict__ sim_b, const float* __restrict__ thr,
    float* __restrict__ red) {
  const int t = blockIdx.x, tid = threadIdx.x;
  const int b = t >> 10, s = t & (Ss - 1);
  const float* row = sim_b + ((size_t)b * Ss + s) * Ss;
  const float th = thr[b];
  float cnt = 0.0f;
  for (int j = tid; j < Ss; j += 256) {
    float pen = expf(-fabsf((float)(s - j)) * (1.0f / (Ss * 0.2f)));
    if (row[j] * pen > th) cnt += 1.0f;
  }
  __shared__ float sred[256];
  sred[tid] = cnt; __syncthreads();
  for (int k = 128; k > 0; k >>= 1) { if (tid < k) sred[tid] += sred[tid + k]; __syncthreads(); }
  if (tid == 0) red[t] = sred[0] * (1.0f / Ss);
}

// importance -> compression-factor routing; count tokens per expert
__global__ __launch_bounds__(256) void route_kernel(
    const float* __restrict__ tokens, const float* __restrict__ w_imp,
    const float* __restrict__ b_imp, const float* __restrict__ red,
    int* __restrict__ fidx, int* __restrict__ counts) {
  const int t = blockIdx.x, tid = threadIdx.x;
  const float* row = tokens + (size_t)t * Dd;
  float p = 0.0f;
  for (int d = tid; d < Dd; d += 256) p += row[d] * w_imp[d];
  __shared__ float sred[256];
  sred[tid] = p; __syncthreads();
  for (int k = 128; k > 0; k >>= 1) { if (tid < k) sred[tid] += sred[tid + k]; __syncthreads(); }
  if (tid == 0) {
    float importance = 1.0f / (1.0f + expf(-(sred[0] + b_imp[0])));
    float imp = importance * (1.0f - red[t]);
    float x = fminf(fmaxf(1.0f - imp, 0.0f), 1.0f);
    int f = (int)rintf(x * (Ff - 1));
    f = f < 0 ? 0 : (f > Ff - 1 ? Ff - 1 : f);
    fidx[t] = f;
    atomicAdd(&counts[f], 1);
  }
}

__global__ void seg_kernel(const int* __restrict__ counts, int* __restrict__ segOff,
                           int* __restrict__ cursor) {
  if (threadIdx.x == 0 && blockIdx.x == 0) {
    int a = 0;
    for (int f = 0; f < Ff; ++f) { segOff[f] = a; a += counts[f]; cursor[f] = 0; }
    segOff[Ff] = a;
  }
}

__global__ void scatter_kernel(const int* __restrict__ fidx, const int* __restrict__ segOff,
                               int* __restrict__ cursor, int* __restrict__ perm) {
  int t = blockIdx.x * blockDim.x + threadIdx.x;
  if (t < Tt) {
    int f = fidx[t];
    int pos = atomicAdd(&cursor[f], 1);
    perm[segOff[f] + pos] = t;
  }
}

// LayerNorm on compact expert output + scatter back to [B,S,D] token order
__global__ __launch_bounds__(256) void ln_scatter_kernel(
    const float* __restrict__ Zf, const int* __restrict__ segOff,
    const int* __restrict__ perm, const float* __restrict__ ln_g,
    const float* __restrict__ ln_b, float* __restrict__ recon) {
  const int i = blockIdx.x, tid = threadIdx.x;
  int e = 0;
#pragma unroll
  for (int f = 0; f < Ff - 1; ++f) if (i >= segOff[f + 1]) e = f + 1;
  const int token = perm[i];
  const float* z = Zf + (size_t)i * Dd;
  __shared__ float s1[256], s2[256];
  float a = 0.0f, b = 0.0f;
  for (int d = tid; d < Dd; d += 256) { float v = z[d]; a += v; b += v * v; }
  s1[tid] = a; s2[tid] = b; __syncthreads();
  for (int k = 128; k > 0; k >>= 1) {
    if (tid < k) { s1[tid] += s1[tid + k]; s2[tid] += s2[tid + k]; }
    __syncthreads();
  }
  const float m   = s1[0] * (1.0f / Dd);
  const float var = s2[0] * (1.0f / Dd) - m * m;
  const float inv = rsqrtf(var + 1e-5f);
  float* out = recon + (size_t)token * Dd;
  for (int d = tid; d < Dd; d += 256)
    out[d] = (z[d] - m) * inv * ln_g[e * Dd + d] + ln_b[e * Dd + d];
}

// gated = where(gate>0.1, gate*res, 0); also build bf16 concat [recon | gated]
__global__ void gate_concat_kernel(const float* __restrict__ recon,
                                   const float* __restrict__ res,
                                   const float* __restrict__ gate,
                                   float* __restrict__ gated,
                                   bf16* __restrict__ concat) {
  size_t i = (size_t)blockIdx.x * blockDim.x + threadIdx.x;
  if (i < (size_t)Tt * Dd) {
    size_t row = i >> 10, d = i & (Dd - 1);
    float g  = gate[i];
    float gv = (g > 0.1f) ? g * res[i] : 0.0f;
    gated[i] = gv;
    concat[row * (2 * Dd) + d]      = (bf16)recon[i];
    concat[row * (2 * Dd) + Dd + d] = (bf16)gv;
  }
}

// ---------------------------------------------------------------------------
extern "C" void kernel_launch(void* const* d_in, const int* in_sizes, int n_in,
                              void* d_out, int out_size, void* d_ws, size_t ws_size,
                              hipStream_t stream) {
  const float* tokens = (const float*)d_in[0];
  const float* w_tm   = (const float*)d_in[1];
  const float* b_tm   = (const float*)d_in[2];
  const float* w_imp  = (const float*)d_in[3];
  const float* b_imp  = (const float*)d_in[4];
  const float* Wc     = (const float*)d_in[5];
  const float* bc     = (const float*)d_in[6];
  const float* W1     = (const float*)d_in[7];
  const float* b1     = (const float*)d_in[8];
  const float* W2     = (const float*)d_in[9];
  const float* b2     = (const float*)d_in[10];
  const float* ln_g   = (const float*)d_in[11];
  const float* ln_b   = (const float*)d_in[12];
  const float* wr1    = (const float*)d_in[13];
  const float* br1    = (const float*)d_in[14];
  const float* wr2    = (const float*)d_in[15];
  const float* br2    = (const float*)d_in[16];
  const float* wg1    = (const float*)d_in[17];
  const float* bg1    = (const float*)d_in[18];
  const float* wg2    = (const float*)d_in[19];
  const float* bg2    = (const float*)d_in[20];
  const float* we1    = (const float*)d_in[21];
  const float* be1    = (const float*)d_in[22];
  const float* we2    = (const float*)d_in[23];
  const float* be2    = (const float*)d_in[24];
  float* out = (float*)d_out;

  char* ws = (char*)d_ws;
  size_t off = 0;
  auto alloc = [&](size_t bytes) -> void* {
    size_t a = (off + 255) & ~(size_t)255;
    off = a + bytes;
    return (void*)(ws + a);
  };

  bf16*  tok_bf  = (bf16*)alloc((size_t)Tt * Dd * 2);
  bf16*  nrm_bf  = (bf16*)alloc((size_t)Tt * Dd * 2);
  float* sim_b   = (float*)alloc((size_t)Bb * Ss * Ss * 4);
  unsigned* hist = (unsigned*)alloc(NBINS * 4);
  float* ctx     = (float*)alloc((size_t)Bb * Dd * 4);
  float* thr     = (float*)alloc(Bb * 4);
  float* red     = (float*)alloc(Tt * 4);
  int*   fidx    = (int*)alloc(Tt * 4);
  int*   counts  = (int*)alloc(Ff * 4);
  int*   segOff  = (int*)alloc((Ff + 1) * 4);
  int*   cursor  = (int*)alloc(Ff * 4);
  int*   perm    = (int*)alloc(Tt * 4);
  bf16*  Wc_bf   = (bf16*)alloc((size_t)Ff * Dd * CDm * 2);
  bf16*  W1_bf   = (bf16*)alloc((size_t)Ff * CDm * Hh * 2);
  bf16*  W2_bf   = (bf16*)alloc((size_t)Ff * Hh * Dd * 2);
  bf16*  wr1_bf  = (bf16*)alloc((size_t)Dd * Rr * 2);
  bf16*  wr2_bf  = (bf16*)alloc((size_t)Rr * Dd * 2);
  bf16*  wg1_bf  = (bf16*)alloc((size_t)Dd * Rr * 2);
  bf16*  wg2_bf  = (bf16*)alloc((size_t)Rr * Dd * 2);
  bf16*  we1_bf  = (bf16*)alloc((size_t)2 * Dd * Dd * 2);
  bf16*  we2_bf  = (bf16*)alloc((size_t)Dd * Dd * 2);
  bf16*  C_bf    = (bf16*)alloc((size_t)Tt * CDm * 2);
  bf16*  Hm_bf   = (bf16*)alloc((size_t)Tt * Hh * 2);
  float* Zf      = (float*)alloc((size_t)Tt * Dd * 4);
  float* recon   = (float*)alloc((size_t)Tt * Dd * 4);
  bf16*  r1_bf   = (bf16*)alloc((size_t)Tt * Rr * 2);
  bf16*  g1_bf   = (bf16*)alloc((size_t)Tt * Rr * 2);
  float* res_f   = (float*)alloc((size_t)Tt * Dd * 4);
  float* gate_f  = (float*)alloc((size_t)Tt * Dd * 4);
  float* gated_f = (float*)alloc((size_t)Tt * Dd * 4);
  bf16*  cat_bf  = (bf16*)alloc((size_t)Tt * 2 * Dd * 2);
  bf16*  e1_bf   = (bf16*)alloc((size_t)Tt * Dd * 2);

  // --- 0) convert all GEMM weights to bf16 (inputs must not be mutated) ---
  auto conv = [&](const float* s, bf16* d, long long n) {
    f2bf_kernel<<<(unsigned)((n + 255) / 256), 256, 0, stream>>>(s, d, n);
  };
  conv(Wc,  Wc_bf,  (long long)Ff * Dd * CDm);
  conv(W1,  W1_bf,  (long long)Ff * CDm * Hh);
  conv(W2,  W2_bf,  (long long)Ff * Hh * Dd);
  conv(wr1, wr1_bf, (long long)Dd * Rr);
  conv(wr2, wr2_bf, (long long)Rr * Dd);
  conv(wg1, wg1_bf, (long long)Dd * Rr);
  conv(wg2, wg2_bf, (long long)Rr * Dd);
  conv(we1, we1_bf, (long long)2 * Dd * Dd);
  conv(we2, we2_bf, (long long)Dd * Dd);

  // --- 1) per-token normalization + bf16 tokens; batch context means ---
  norm_kernel<<<Tt, 256, 0, stream>>>(tokens, tok_bf, nrm_bf);
  ctx_kernel<<<dim3(Dd / 256, Bb), 256, 0, stream>>>(tokens, ctx);

  // --- 2) pairwise sims (WMMA) + histogram + diag-block store ---
  hipMemsetAsync(hist,   0, NBINS * sizeof(unsigned), stream);
  hipMemsetAsync(counts, 0, Ff * sizeof(int), stream);
  sims_kernel<<<dim3(Tt / BM, Tt / BN), 256, 0, stream>>>(nrm_bf, sim_b, hist);

  // --- 3) quantile -> per-batch threshold; redundancy; routing ---
  thr_kernel<<<1, 256, 0, stream>>>(hist, ctx, w_tm, b_tm, thr);
  red_kernel<<<Tt, 256, 0, stream>>>(sim_b, thr, red);
  route_kernel<<<Tt, 256, 0, stream>>>(tokens, w_imp, b_imp, red, fidx, counts);
  seg_kernel<<<1, 1, 0, stream>>>(counts, segOff, cursor);
  scatter_kernel<<<Tt / 256, 256, 0, stream>>>(fidx, segOff, cursor, perm);

  // --- 4) routed experts: only the selected expert per token (7x FLOP cut) ---
  expert_gemm_kernel<0, true><<<dim3(Tt / BM, CDm / BN, Ff), 256, 0, stream>>>(
      tok_bf, Wc_bf, bc, nullptr, C_bf, segOff, perm, CDm, Dd, (long)Dd * CDm, CDm);
  expert_gemm_kernel<1, false><<<dim3(Tt / BM, Hh / BN, Ff), 256, 0, stream>>>(
      C_bf, W1_bf, b1, nullptr, Hm_bf, segOff, perm, Hh, CDm, (long)CDm * Hh, Hh);
  expert_gemm_kernel<0, false><<<dim3(Tt / BM, Dd / BN, Ff), 256, 0, stream>>>(
      Hm_bf, W2_bf, b2, Zf, nullptr, segOff, perm, Dd, Hh, (long)Hh * Dd, Dd);
  ln_scatter_kernel<<<Tt, 256, 0, stream>>>(Zf, segOff, perm, ln_g, ln_b, recon);

  // --- 5) residual compressor / gate / enhancer (dense WMMA GEMMs) ---
  gemm_kernel<1><<<dim3(Tt / BM, Rr / BN), 256, 0, stream>>>(
      tok_bf, wr1_bf, br1, nullptr, r1_bf, nullptr, nullptr, Tt, Rr, Dd);
  gemm_kernel<0><<<dim3(Tt / BM, Dd / BN), 256, 0, stream>>>(
      r1_bf, wr2_bf, br2, res_f, nullptr, nullptr, nullptr, Tt, Dd, Rr);
  gemm_kernel<1><<<dim3(Tt / BM, Rr / BN), 256, 0, stream>>>(
      tok_bf, wg1_bf, bg1, nullptr, g1_bf, nullptr, nullptr, Tt, Rr, Dd);
  gemm_kernel<2><<<dim3(Tt / BM, Dd / BN), 256, 0, stream>>>(
      g1_bf, wg2_bf, bg2, gate_f, nullptr, nullptr, nullptr, Tt, Dd, Rr);
  gate_concat_kernel<<<(unsigned)(((size_t)Tt * Dd + 255) / 256), 256, 0, stream>>>(
      recon, res_f, gate_f, gated_f, cat_bf);
  gemm_kernel<1><<<dim3(Tt / BM, Dd / BN), 256, 0, stream>>>(
      cat_bf, we1_bf, be1, nullptr, e1_bf, nullptr, nullptr, Tt, Dd, 2 * Dd);
  // final: out = recon + gated + (e1 @ we2 + be2)
  gemm_kernel<0><<<dim3(Tt / BM, Dd / BN), 256, 0, stream>>>(
      e1_bf, we2_bf, be2, out, nullptr, recon, gated_f, Tt, Dd, Dd);

  (void)in_sizes; (void)n_in; (void)out_size; (void)ws_size;
}
